// HFMiMoV2MoEGate_61546881352282
// MI455X (gfx1250) — matrix operand order repaired and checked by reading, not compiled, not run on hardware
//
#include <hip/hip_runtime.h>
#include <hip/hip_bf16.h>
#include <math.h>

typedef __attribute__((ext_vector_type(16))) __bf16 v16bf;
typedef __attribute__((ext_vector_type(4)))  __bf16 v4bf;
typedef __attribute__((ext_vector_type(8)))  float  v8f;

#define TOP_K       8
#define N_GROUP     8
#define TOPK_GROUP  4
#define E_TOT       256
#define GSIZE       32
#define BM          16      // rows per block
#define KS          64      // K staged per barrier iteration (2 WMMA k-steps)
#define LDAF        68      // padded LDS row stride (floats): 68*4B -> 4-bank row skew, 16B aligned

// ---------------------------------------------------------------------------
// Kernel 0: one-time weight f32 -> bf16 conversion into workspace (2 MB)
// ---------------------------------------------------------------------------
__global__ void convert_weight_bf16(const float* __restrict__ w,
                                    __bf16* __restrict__ o, int total)
{
    int i = (blockIdx.x * blockDim.x + threadIdx.x) * 4;
    if (i < total) {
        float4 v = *(const float4*)(w + i);
        v4bf b;
        b[0] = (__bf16)v.x; b[1] = (__bf16)v.y;
        b[2] = (__bf16)v.z; b[3] = (__bf16)v.w;
        *(v4bf*)(o + i) = b;
    }
}

// ---------------------------------------------------------------------------
// Kernel 1: fused gate GEMM (bf16 WMMA, f32 accum) + group-limited top-k
// ---------------------------------------------------------------------------
template <bool B_BF16>
__global__ __launch_bounds__(256, 2)
void moe_gate_fused_kernel(const float* __restrict__ hs,   // [n, h] f32
                           const void*  __restrict__ wB,   // [E, h] bf16 (ws) or f32
                           const float* __restrict__ bias, // [E]
                           int*   __restrict__ idx_out,    // [n, 8]
                           float* __restrict__ w_out,      // [n, 8]
                           int n, int h)
{
    __shared__ float ldsA[BM * LDAF];          // async-staged f32 A tile (16 x 64)
    __shared__ float ldsLogits[BM * E_TOT];    // fused epilogue scratch (16 KB)

    const int tid  = threadIdx.x;
    const int lane = tid & 31;                 // wave32
    const int wave = tid >> 5;                 // 8 waves
    const int row0 = blockIdx.x * BM;

    const int m   = lane & 15;                 // M (A) / N (B) index within 16x16 tile
    const int kbA = (lane < 16) ? 0 : 16;      // A frag K base: 16 contiguous K
    const int kbB = (lane < 16) ? 0 : 16;      // B frag K base: 16 contiguous K

    const int eb0 = wave * 32;
    const int eb1 = eb0 + 16;
    const __bf16* wb = (const __bf16*)wB;
    const float*  wf = (const float*)wB;

    v8f acc0 = {}; v8f acc1 = {};

    // async A staging: 16 rows x 64 f32 = 1024 floats; 4 per thread (b128)
    const int aIdx = tid * 4;
    const int ar = aIdx >> 6;                  // 0..15
    const int ac = aIdx & 63;                  // multiple of 4
    const float* aSrc = hs + (size_t)(row0 + ar) * h + ac;
    const unsigned aDst = (unsigned)(unsigned long long)&ldsA[ar * LDAF + ac];

    for (int kk = 0; kk < h; kk += KS) {
        // ---- stage A tile: async global -> LDS (no VGPR round trip) ----
        asm volatile("global_load_async_to_lds_b128 %0, %1, off"
                     :: "v"(aDst), "v"(aSrc + kk) : "memory");
        asm volatile("s_wait_asynccnt 0" ::: "memory");
        __syncthreads();

        if (kk + KS < h) {   // prefetch next weight chunk toward L2/L0
            if (B_BF16) {
                __builtin_prefetch(wb + (size_t)(eb0 + m) * h + kk + KS, 0, 1);
                __builtin_prefetch(wb + (size_t)(eb1 + m) * h + kk + KS, 0, 1);
            } else {
                __builtin_prefetch(wf + (size_t)(eb0 + m) * h + kk + KS, 0, 1);
                __builtin_prefetch(wf + (size_t)(eb1 + m) * h + kk + KS, 0, 1);
            }
        }

        #pragma unroll
        for (int s = 0; s < 2; ++s) {          // two K=32 WMMA steps per stage
            // ---- A fragment: f32 LDS -> bf16, 16 contiguous K per lane half (k = 16h + j), same order as B ----
            const float* aRow = &ldsA[m * LDAF + s * 32];
            float t[16];
            *(float4*)&t[0]  = *(const float4*)(aRow + kbA);
            *(float4*)&t[4]  = *(const float4*)(aRow + kbA + 4);
            *(float4*)&t[8]  = *(const float4*)(aRow + kbA + 8);
            *(float4*)&t[12] = *(const float4*)(aRow + kbA + 12);
            v16bf af;
            #pragma unroll
            for (int j = 0; j < 16; ++j) af[j] = (__bf16)t[j];

            // ---- B fragments: lane = expert row, 16 contiguous K ----
            v16bf bf0, bf1;
            const size_t off = (size_t)kk + s * 32 + kbB;
            if (B_BF16) {
                bf0 = *(const v16bf*)(wb + (size_t)(eb0 + m) * h + off);
                bf1 = *(const v16bf*)(wb + (size_t)(eb1 + m) * h + off);
            } else {
                const float* b0 = wf + (size_t)(eb0 + m) * h + off;
                const float* b1 = wf + (size_t)(eb1 + m) * h + off;
                #pragma unroll
                for (int j = 0; j < 16; ++j) {
                    bf0[j] = (__bf16)b0[j];
                    bf1[j] = (__bf16)b1[j];
                }
            }

            acc0 = __builtin_amdgcn_wmma_f32_16x16x32_bf16(false, af, false, bf0,
                                                           (short)0, acc0, false, false);
            acc1 = __builtin_amdgcn_wmma_f32_16x16x32_bf16(false, af, false, bf1,
                                                           (short)0, acc1, false, false);
        }
        __syncthreads();   // protect ldsA before next stage overwrites
    }

    // ---- scatter logits per C layout: VGPR i -> row i (lanes 0-15) / i+8 (16-31) ----
    {
        const int rlo = (lane < 16) ? 0 : 8;
        #pragma unroll
        for (int i = 0; i < 8; ++i) {
            ldsLogits[(i + rlo) * E_TOT + eb0 + m] = acc0[i];
            ldsLogits[(i + rlo) * E_TOT + eb1 + m] = acc1[i];
        }
    }
    __syncthreads();

    // ---- routing epilogue: one thread per row ----
    if (tid < BM) {
        const int row = row0 + tid;
        const float* lg = &ldsLogits[tid * E_TOT];

        float gs[N_GROUP];
        for (int g = 0; g < N_GROUP; ++g) {
            float m1 = -INFINITY, m2 = -INFINITY;
            const int base = g * GSIZE;
            for (int j = 0; j < GSIZE; ++j) {
                float s = 1.0f / (1.0f + __expf(-lg[base + j])) + bias[base + j];
                if (s > m1) { m2 = m1; m1 = s; }
                else if (s > m2) { m2 = s; }
            }
            gs[g] = m1 + m2;
        }

        unsigned gmask = 0;
        for (int k = 0; k < TOPK_GROUP; ++k) {
            float best = -INFINITY; int bi = 0;
            for (int g = 0; g < N_GROUP; ++g)
                if (!((gmask >> g) & 1u) && gs[g] > best) { best = gs[g]; bi = g; }
            gmask |= 1u << bi;
        }

        float tv[TOP_K]; int ti[TOP_K];
        for (int k = 0; k < TOP_K; ++k) { tv[k] = -INFINITY; ti[k] = 0; }
        for (int g = 0; g < N_GROUP; ++g) {
            if (!((gmask >> g) & 1u)) continue;
            const int base = g * GSIZE;
            for (int j = 0; j < GSIZE; ++j) {
                float s = 1.0f / (1.0f + __expf(-lg[base + j])) + bias[base + j];
                if (s > tv[TOP_K - 1]) {
                    int p = TOP_K - 1;
                    while (p > 0 && s > tv[p - 1]) { tv[p] = tv[p-1]; ti[p] = ti[p-1]; --p; }
                    tv[p] = s; ti[p] = base + j;
                }
            }
        }

        float tw[TOP_K]; float wsum = 0.f;
        for (int k = 0; k < TOP_K; ++k) {
            float s = 1.0f / (1.0f + __expf(-lg[ti[k]]));
            tw[k] = s; wsum += s;
        }
        const float scale = 2.5f / (wsum + 1e-20f);
        for (int k = 0; k < TOP_K; ++k) {
            idx_out[(size_t)row * TOP_K + k] = ti[k];
            w_out[(size_t)row * TOP_K + k]   = tw[k] * scale;
        }
    }
}

extern "C" void kernel_launch(void* const* d_in, const int* in_sizes, int n_in,
                              void* d_out, int out_size, void* d_ws, size_t ws_size,
                              hipStream_t stream) {
    const float* hs   = (const float*)d_in[0];
    const float* wgt  = (const float*)d_in[1];
    const float* bias = (const float*)d_in[2];

    const int E = in_sizes[2];                  // 256
    const int h = in_sizes[1] / E;              // 4096
    const int n = in_sizes[0] / h;              // 16384

    int*   idx_out = (int*)d_out;
    float* w_out   = (float*)d_out + (size_t)n * TOP_K;

    dim3 grid(n / BM);
    dim3 block(256);

    const size_t wbytes = (size_t)E * h * sizeof(__bf16);
    if (ws_size >= wbytes) {
        // one-time (per launch) weight f32->bf16 into workspace: 2 MB, trivial cost
        int total = E * h;
        convert_weight_bf16<<<dim3(total / 4 / 256), block, 0, stream>>>(wgt, (__bf16*)d_ws, total);
        moe_gate_fused_kernel<true><<<grid, block, 0, stream>>>(hs, d_ws, bias, idx_out, w_out, n, h);
    } else {
        moe_gate_fused_kernel<false><<<grid, block, 0, stream>>>(hs, wgt, bias, idx_out, w_out, n, h);
    }
}
